// EntRelJointDecoder_68212670595943
// MI455X (gfx1250) — compile-verified, matrix operand order checked by reading, not run on hardware
//
#include <hip/hip_runtime.h>
#include <math.h>

// ---------- problem constants ----------
#define BB 4
#define SS 96
#define HH 768
#define MM 256
#define VV 20
#define OO 20
#define NROW (BB * SS * SS)      // 36864 pair rows
#define NBS  (BB * SS)           // 384 token rows
#define NCOL (SS * OO)           // 1920 q columns per batch

// ---------- CDNA5 vector types ----------
typedef __attribute__((ext_vector_type(16))) __bf16 v16bf;
typedef __attribute__((ext_vector_type(8)))  float  v8f;
typedef __attribute__((ext_vector_type(4)))  float  v4f;

typedef __attribute__((address_space(3))) unsigned short lds_ushort;

// ---------- workspace layout (bytes) ----------
#define OFF_XH    0u            // 384*256 f32
#define OFF_XT    393216u       // 384*256 f32
#define OFF_VALB  786432u       // 384*256 bf16
#define OFF_UB    983040u       // 5120*256 bf16
#define OFF_WFT   3604480u      // 32*256 bf16 (padded final_W^T)
#define OFF_UVB   3620864u      // 384*5120 bf16
#define OFF_PAIR  7553024u      // 36864*256 bf16
#define OFF_ACC   26427392u     // 4 f32 accumulators

__device__ __forceinline__ float gelu_exact(float x) {
    return 0.5f * x * (1.0f + erff(x * 0.70710678118654752f));
}

__device__ __forceinline__ unsigned short f2bf(float x) {
    unsigned u = __float_as_uint(x);
    unsigned r = u + 0x7FFFu + ((u >> 16) & 1u);
    return (unsigned short)(r >> 16);
}

// Load one 16x32 bf16 WMMA fragment from a row-major K-contiguous panel.
// Per ISA 7.12.2: lane L%16 = row (or col), L/16 selects interleaved K-half:
//   regs 0..3 <- K[(L/16)*8 .. +7], regs 4..7 <- K[16+(L/16)*8 .. +7]
__device__ __forceinline__ v16bf load_frag(const unsigned short* base, int ldk,
                                           int k0, int lane) {
    int row = lane & 15;
    int sel = lane >> 4;
    const unsigned short* p = base + (long)row * ldk + k0 + sel * 8;
    v4f lo = *reinterpret_cast<const v4f*>(p);
    v4f hi = *reinterpret_cast<const v4f*>(p + 16);
    v16bf f;
    __builtin_memcpy(&f, &lo, 16);
    __builtin_memcpy(reinterpret_cast<char*>(&f) + 16, &hi, 16);
    return f;
}

#define WMMA_BF16(a, b, c) \
    __builtin_amdgcn_wmma_f32_16x16x32_bf16(false, (a), false, (b), (short)0, (c), false, false)

// ---------------------------------------------------------------------------
// init accumulators
__global__ void k_init(float* acc) {
    if (threadIdx.x < 4) acc[threadIdx.x] = 0.0f;
}

// Xh = x @ W_head, Xt = x @ W_tail, value = GELU(x @ value_W + b) (bf16)
__global__ __launch_bounds__(256) void k_prep(const float* __restrict__ x,
                                              const float* __restrict__ pairW,
                                              const float* __restrict__ valueW,
                                              const float* __restrict__ value_b,
                                              float* __restrict__ Xh,
                                              float* __restrict__ Xt,
                                              unsigned short* __restrict__ valb) {
    __shared__ float xs[HH];
    int r = blockIdx.x;                       // token row (b*S+t)
    const float* xr = x + (long)r * HH;
    for (int k = threadIdx.x; k < HH; k += 256) xs[k] = xr[k];
    __syncthreads();
    int m = threadIdx.x;                      // output feature 0..255
    float ah = 0.f, at = 0.f, av = 0.f;
    for (int k = 0; k < HH; ++k) {
        float xv = xs[k];
        ah = fmaf(xv, pairW[(long)k * MM + m], ah);
        at = fmaf(xv, pairW[(long)(HH + k) * MM + m], at);
        av = fmaf(xv, valueW[(long)k * MM + m], av);
    }
    Xh[(long)r * MM + m] = ah;
    Xt[(long)r * MM + m] = at;
    valb[(long)r * MM + m] = f2bf(gelu_exact(av + value_b[m]));
}

// U [O,M,M] f32 -> bf16 (natural layout == B^T panel rows (o,i) x K=j)
__global__ __launch_bounds__(256) void k_convU(const float* __restrict__ U,
                                               unsigned short* __restrict__ Ub, int n) {
    for (int i = blockIdx.x * 256 + threadIdx.x; i < n; i += gridDim.x * 256)
        Ub[i] = f2bf(U[i]);
}

// final_W [M,V] -> padded transposed bf16 [32][256]
__global__ __launch_bounds__(256) void k_wft(const float* __restrict__ fW,
                                             unsigned short* __restrict__ Wft) {
    int i = blockIdx.x * 256 + threadIdx.x;
    if (i < 32 * MM) {
        int n = i / MM, k = i % MM;
        float v = (n < VV) ? fW[(long)k * VV + n] : 0.0f;
        Wft[(long)n * MM + k] = f2bf(v);
    }
}

// pair[b,x,y,m] = bf16(GELU(Xh[b,x,m] + Xt[b,y,m] + pair_b[m]))
__global__ __launch_bounds__(256) void k_pairk(const float* __restrict__ Xh,
                                               const float* __restrict__ Xt,
                                               const float* __restrict__ pair_b,
                                               unsigned short* __restrict__ pairb) {
    const int total = NROW * MM;
    for (int i = blockIdx.x * 256 + threadIdx.x; i < total; i += gridDim.x * 256) {
        int m = i & (MM - 1);
        int r = i >> 8;
        int b = r / (SS * SS);
        int rem = r - b * SS * SS;
        int xx = rem / SS, yy = rem - xx * SS;
        float v = Xh[((long)(b * SS + xx)) * MM + m] +
                  Xt[((long)(b * SS + yy)) * MM + m] + pair_b[m];
        pairb[i] = f2bf(gelu_exact(v));
    }
}

// uv[(b,z)][(o,i)] = sum_j value[(b,z)][j] * U[(o,i)][j]  — WMMA bf16 GEMM
__global__ __launch_bounds__(256, 1) void k_uv(const unsigned short* __restrict__ valb,
                                               const unsigned short* __restrict__ Ub,
                                               unsigned short* __restrict__ uvb) {
    int wv = threadIdx.x >> 5, lane = threadIdx.x & 31;
    int w = blockIdx.x * 8 + wv;              // 0..1919
    int rt = w / 80, colg = w % 80;           // 24 row tiles x 80 col groups(64)
    const unsigned short* A = valb + (long)rt * 16 * MM;
    v16bf a[8];
#pragma unroll
    for (int kb = 0; kb < 8; ++kb) a[kb] = load_frag(A, MM, kb * 32, lane);
    int sel = lane >> 4, c = lane & 15;
#pragma unroll
    for (int t = 0; t < 4; ++t) {
        int c0 = colg * 64 + t * 16;
        const unsigned short* Bm = Ub + (long)c0 * MM;
        v16bf bfr[8];
#pragma unroll
        for (int kb = 0; kb < 8; ++kb) bfr[kb] = load_frag(Bm, MM, kb * 32, lane);
        v8f acc = {0.f, 0.f, 0.f, 0.f, 0.f, 0.f, 0.f, 0.f};
#pragma unroll
        for (int kb = 0; kb < 8; ++kb) acc = WMMA_BF16(a[kb], bfr[kb], acc);
        int col = c0 + c;
#pragma unroll
        for (int r = 0; r < 8; ++r) {
            int rowg = rt * 16 + r + sel * 8;
            uvb[(long)rowg * (OO * MM) + col] = f2bf(acc[r]);
        }
    }
}

// joint_score = pair @ final_W + b, fused masked CE  (WMMA, 16 rows/wave)
__global__ __launch_bounds__(256) void k_joint(const unsigned short* __restrict__ pairb,
                                               const unsigned short* __restrict__ Wft,
                                               const float* __restrict__ final_b,
                                               const int* __restrict__ jl,
                                               const unsigned char* __restrict__ jmask,
                                               float* __restrict__ accum) {
    __shared__ float lds[8 * 16 * 32];        // 16 KB, one 16x32 tile per wave
    int wv = threadIdx.x >> 5, lane = threadIdx.x & 31;
    long w = (long)blockIdx.x * 8 + wv;       // 0..2303
    long R0 = w * 16;
    const unsigned short* A = pairb + R0 * MM;
    float* L = lds + wv * 512;
    v16bf b0[8], b1[8];
#pragma unroll
    for (int kb = 0; kb < 8; ++kb) {
        b0[kb] = load_frag(Wft, MM, kb * 32, lane);
        b1[kb] = load_frag(Wft + 16 * MM, MM, kb * 32, lane);
    }
    v8f acc0 = {0.f, 0.f, 0.f, 0.f, 0.f, 0.f, 0.f, 0.f};
    v8f acc1 = {0.f, 0.f, 0.f, 0.f, 0.f, 0.f, 0.f, 0.f};
#pragma unroll
    for (int kb = 0; kb < 8; ++kb) {
        v16bf a = load_frag(A, MM, kb * 32, lane);
        acc0 = WMMA_BF16(a, b0[kb], acc0);
        acc1 = WMMA_BF16(a, b1[kb], acc1);
    }
    int sel = lane >> 4, c = lane & 15;
#pragma unroll
    for (int r = 0; r < 8; ++r) {
        L[(r + sel * 8) * 32 + c] = acc0[r];
        L[(r + sel * 8) * 32 + 16 + c] = acc1[r];
    }
    float nll = 0.f, cnt = 0.f;
    if (lane < 16) {
        int row = lane;
        long gr = R0 + row;
        int lab = jl[gr];
        float mk = jmask[gr] ? 1.0f : 0.0f;
        float mx = -1e30f;
        for (int o = 0; o < VV; ++o) mx = fmaxf(mx, L[row * 32 + o] + final_b[o]);
        float s = 0.f;
        for (int o = 0; o < VV; ++o) s += __expf(L[row * 32 + o] + final_b[o] - mx);
        float lse = mx + __logf(s);
        nll = (lse - (L[row * 32 + lab] + final_b[lab])) * mk;
        cnt = mk;
    }
    for (int off = 16; off > 0; off >>= 1) {
        nll += __shfl_down(nll, off, 32);
        cnt += __shfl_down(cnt, off, 32);
    }
    if (lane == 0) { atomicAdd(&accum[0], nll); atomicAdd(&accum[1], cnt); }
}

// DOMINANT: q_score GEMM (pair x uv^T) fused with CE(softmax(q)) loss.
// Block = 128 rows (8 waves x 16) x 240 cols: B panel SHARED by all waves,
// staged to LDS in 80col x 64K chunks with CDNA5 async-to-LDS copies
// (ASYNCcnt), fragments then served by ds_load_b128. Accumulators (5 x v8f)
// persist in VGPRs across the 4 K-chunks.
#define BST_LDK 72   // padded LDS row stride (elems): 144B, 16B-aligned, bank-spread
__global__ __launch_bounds__(256, 1) void k_qloss(const unsigned short* __restrict__ pairb,
                                                  const unsigned short* __restrict__ uvb,
                                                  const int* __restrict__ quint,
                                                  const unsigned char* __restrict__ qmask,
                                                  float* __restrict__ accum) {
    __shared__ float cstage[8 * 16 * 80];            // 40 KB per-wave C panels
    __shared__ unsigned short bstage[80 * BST_LDK];  // 11.25 KB shared B chunk
    int wv = threadIdx.x >> 5, lane = threadIdx.x & 31;
    int rb = blockIdx.x >> 3;                 // 0..287 row blocks of 128
    int cp = blockIdx.x & 7;                  // 0..7 column panels of 240
    long R0w = (long)rb * 128 + wv * 16;      // this wave's 16 rows
    int b = rb / 72;                          // 9216/128 = 72 row-blocks/batch
    const unsigned short* A = pairb + R0w * MM;
    v16bf a[8];
#pragma unroll
    for (int kb = 0; kb < 8; ++kb) a[kb] = load_frag(A, MM, kb * 32, lane);
    const unsigned short* Bb = uvb + (long)b * NCOL * MM;
    float* L = cstage + wv * 1280;
    int NB = cp * 240;
    int sel = lane >> 4, c = lane & 15;
    unsigned bst_base = (unsigned)(unsigned long)(lds_ushort*)bstage;
    float nll = 0.f, cnt = 0.f;
    for (int sub = 0; sub < 3; ++sub) {
        int p0 = NB + sub * 80;               // 80 global cols = 4 z-groups
        v8f acc[5];
#pragma unroll
        for (int t = 0; t < 5; ++t)
            acc[t] = (v8f){0.f, 0.f, 0.f, 0.f, 0.f, 0.f, 0.f, 0.f};
        for (int kc = 0; kc < 4; ++kc) {      // K chunks of 64
            __syncthreads();                  // previous chunk fully consumed
            // cooperative async stage: 80 cols x 64 K bf16 (640 x 16B units)
            {
                int u = threadIdx.x;
#pragma unroll
                for (int it = 0; it < 3; ++it, u += 256) {
                    if (u < 640) {
                        int col = u >> 3, part = u & 7;
                        const unsigned short* g =
                            Bb + (long)(p0 + col) * MM + kc * 64 + part * 8;
                        unsigned l = bst_base +
                                     (unsigned)((col * BST_LDK + part * 8) * 2);
                        asm volatile("global_load_async_to_lds_b128 %0, %1, off"
                                     :: "v"(l), "v"(g) : "memory");
                    }
                }
            }
            asm volatile("s_wait_asynccnt 0x0" ::: "memory");
            __syncthreads();                  // chunk visible to all waves
            // 5 tiles x 2 K-fragments from LDS, WMMA accumulate
#pragma unroll
            for (int t = 0; t < 5; ++t) {
                const unsigned short* Bp = bstage + (t * 16 + c) * BST_LDK;
#pragma unroll
                for (int kb2 = 0; kb2 < 2; ++kb2) {
                    const unsigned short* p = Bp + kb2 * 32 + sel * 8;
                    v4f lo = *reinterpret_cast<const v4f*>(p);
                    v4f hi = *reinterpret_cast<const v4f*>(p + 16);
                    v16bf bf;
                    __builtin_memcpy(&bf, &lo, 16);
                    __builtin_memcpy(reinterpret_cast<char*>(&bf) + 16, &hi, 16);
                    acc[t] = WMMA_BF16(a[kc * 2 + kb2], bf, acc[t]);
                }
            }
        }
        // stage this wave's 16x80 f32 panel and fuse the double-softmax loss
#pragma unroll
        for (int t = 0; t < 5; ++t)
#pragma unroll
            for (int r = 0; r < 8; ++r)
                L[(r + sel * 8) * 80 + t * 16 + c] = acc[t][r];
        int z0 = p0 / OO;
#pragma unroll
        for (int it = 0; it < 2; ++it) {
            int p = lane + it * 32;           // 64 (row,z) pairs -> 2 per lane
            int row = p >> 2, zl = p & 3;
            long gr = R0w + row;              // encodes ((b*S+x)*S+y)
            int z = z0 + zl;
            long qi = gr * SS + z;
            int lab = quint[qi];
            float mk = qmask[qi] ? 1.0f : 0.0f;
            const float* Q = L + row * 80 + zl * OO;
            float mx = -1e30f;
            for (int o = 0; o < OO; ++o) mx = fmaxf(mx, Q[o]);
            float s = 0.f;
            for (int o = 0; o < OO; ++o) s += __expf(Q[o] - mx);
            float inv = 1.0f / s;
            float plab = __expf(Q[lab] - mx) * inv;   // softmax at label
            float s2 = 0.f;                           // sum_o exp(p_o)
            for (int o = 0; o < OO; ++o) s2 += __expf(__expf(Q[o] - mx) * inv);
            nll += (__logf(s2) - plab) * mk;
            cnt += mk;
        }
    }
    for (int off = 16; off > 0; off >>= 1) {
        nll += __shfl_down(nll, off, 32);
        cnt += __shfl_down(cnt, off, 32);
    }
    if (lane == 0) { atomicAdd(&accum[2], nll); atomicAdd(&accum[3], cnt); }
}

__global__ void k_final(const float* __restrict__ acc, float* __restrict__ out) {
    if (threadIdx.x == 0)
        out[0] = acc[0] / acc[1] + acc[2] / acc[3];
}

// ---------------------------------------------------------------------------
extern "C" void kernel_launch(void* const* d_in, const int* in_sizes, int n_in,
                              void* d_out, int out_size, void* d_ws, size_t ws_size,
                              hipStream_t stream) {
    (void)in_sizes; (void)n_in; (void)out_size; (void)ws_size;
    const float* x        = (const float*)d_in[0];
    const float* pairW    = (const float*)d_in[1];
    const float* pair_b   = (const float*)d_in[2];
    const float* finalW   = (const float*)d_in[3];
    const float* final_b  = (const float*)d_in[4];
    const float* valueW   = (const float*)d_in[5];
    const float* value_b  = (const float*)d_in[6];
    const float* U        = (const float*)d_in[7];
    const int*   jl       = (const int*)d_in[8];
    const unsigned char* jm  = (const unsigned char*)d_in[9];
    const int*   qm       = (const int*)d_in[10];
    const unsigned char* qmm = (const unsigned char*)d_in[11];

    char* ws = (char*)d_ws;
    float*          Xh    = (float*)(ws + OFF_XH);
    float*          Xt    = (float*)(ws + OFF_XT);
    unsigned short* valb  = (unsigned short*)(ws + OFF_VALB);
    unsigned short* Ub    = (unsigned short*)(ws + OFF_UB);
    unsigned short* Wft   = (unsigned short*)(ws + OFF_WFT);
    unsigned short* uvb   = (unsigned short*)(ws + OFF_UVB);
    unsigned short* pairb = (unsigned short*)(ws + OFF_PAIR);
    float*          acc   = (float*)(ws + OFF_ACC);
    float*          out   = (float*)d_out;

    k_init <<<1, 32, 0, stream>>>(acc);
    k_prep <<<NBS, 256, 0, stream>>>(x, pairW, valueW, value_b, Xh, Xt, valb);
    k_convU<<<2048, 256, 0, stream>>>(U, Ub, OO * MM * MM);
    k_wft  <<<32, 256, 0, stream>>>(finalW, Wft);
    k_pairk<<<4096, 256, 0, stream>>>(Xh, Xt, pair_b, pairb);
    k_uv   <<<240, 256, 0, stream>>>(valb, Ub, uvb);
    k_joint<<<288, 256, 0, stream>>>(pairb, Wft, final_b, jl, jm, acc);
    k_qloss<<<2304, 256, 0, stream>>>(pairb, uvb, qm, qmm, acc);
    k_final<<<1, 1, 0, stream>>>(acc, out);
}
